// VQImageBERTObj_65893388255593
// MI455X (gfx1250) — matrix operand order repaired and checked by reading, not compiled, not run on hardware
//
#include <hip/hip_runtime.h>

// MI455X / gfx1250: wave32, WMMA (not MFMA), 320KB LDS per WGP.
// Fused GEMM + online logsumexp, bf16x3 compensated precision,
// double-buffered LDS staging + 2-deep register prefetch.

typedef __bf16 bf16;
typedef __bf16 v16bf __attribute__((ext_vector_type(16)));
typedef float  v8f   __attribute__((ext_vector_type(8)));

#define DEV __device__ __forceinline__

DEV v8f wmma_bf16(v16bf a, v16bf b, v8f c) {
  // (neg_a, A, neg_b, B, c_mod, C, reuse_a, reuse_b)
  return __builtin_amdgcn_wmma_f32_16x16x32_bf16(false, a, false, b, (short)0, c, false, false);
}

DEV void split_bf16(float f, bf16& h, bf16& l) {
  h = (bf16)f;                 // RNE to bf16 (high part)
  l = (bf16)(f - (float)h);    // residual (low part)
}

template<int XOR_MASK>
DEV float swz_xor(float x) {
  // ds_swizzle group-of-32: offset = {xor[14:10], or[9:5], and[4:0]}
  constexpr int off = (XOR_MASK << 10) | 0x1f;
  return __int_as_float(__builtin_amdgcn_ds_swizzle(__float_as_int(x), off));
}

DEV void lse_merge(float& m, float& s, float m2, float s2) {
  float mn = fmaxf(m, m2);
  s = s * __expf(m - mn) + s2 * __expf(m2 - mn);
  m = mn;
}

DEV float wave_sum32(float v) {
  v += swz_xor<16>(v);
  v += swz_xor<8>(v);
  v += swz_xor<4>(v);
  v += swz_xor<2>(v);
  v += swz_xor<1>(v);
  return v;
}

// ---------------------------------------------------------------------------
// Fused GEMM + row-wise online logsumexp.
// Pass y==0: rows=X, cols=W, +bias b        -> lseM   (MIM denominator)
// Pass y==1: rows=E, cols=X, no bias        -> lseN   (NCE denominator)
// Block: 256 threads = 8 waves. Panel: 64 rows x all columns.
// LDS: row panel 64x1024 bf16 hi/lo (256KB) + 2x double-buffered 128x32 chunk.
// ---------------------------------------------------------------------------
__global__ __launch_bounds__(256) void lse_gemm_kernel(
    const float* __restrict__ X, const float* __restrict__ Wv,
    const float* __restrict__ bias, float* __restrict__ lseM,
    const float* __restrict__ E, float* __restrict__ lseN,
    int nColsPass0, int nColsPass1)
{
  constexpr int H = 1024;
  constexpr int BCH = 128 * 32;            // B chunk elements
  extern __shared__ __align__(16) char smem[];
  bf16* Ahi = (bf16*)smem;                 // [64][1024]
  bf16* Alo = Ahi + 64 * H;                // [64][1024]
  bf16* Bbuf = Alo + 64 * H;               // [2 bufs][hi|lo][128][32]
  float* part = (float*)(Bbuf + 4 * BCH);  // [8 waves][16 rows][m,s]
  float* run  = part + 8 * 16 * 2;         // [64 rows][m,s]

  const float* Rg; const float* Cg; const float* bp; float* outp; int nCols;
  if (blockIdx.y == 0) { Rg = X; Cg = Wv; bp = bias;   outp = lseM; nCols = nColsPass0; }
  else                 { Rg = E; Cg = X;  bp = nullptr; outp = lseN; nCols = nColsPass1; }

  const int tid  = threadIdx.x;
  const int lane = tid & 31;           // wave32
  const int w    = tid >> 5;           // 8 waves
  const int half = lane >> 4;
  const int l15  = lane & 15;
  const int mt   = w & 3;              // m-tile 0..3 (16 rows each)
  const int ng   = w >> 2;             // n-group 0..1 (4 n-tiles each)
  const int rowBase = blockIdx.x * 64;

  // ---- Stage row panel (64 x 1024 fp32) into LDS as bf16 hi/lo, once ----
  #pragma unroll 1
  for (int i = 0; i < 16; ++i) {
    int c = tid + 256 * i;                 // chunk of 16 floats
    int rowLocal = c >> 6;
    int k = (c & 63) << 4;
    const float4* s4 = (const float4*)(Rg + (size_t)(rowBase + rowLocal) * H + k);
    __align__(16) bf16 h[16]; __align__(16) bf16 l[16];
    #pragma unroll
    for (int q = 0; q < 4; ++q) {
      float4 f = s4[q];
      split_bf16(f.x, h[4*q+0], l[4*q+0]);
      split_bf16(f.y, h[4*q+1], l[4*q+1]);
      split_bf16(f.z, h[4*q+2], l[4*q+2]);
      split_bf16(f.w, h[4*q+3], l[4*q+3]);
    }
    *(uint4*)&Ahi[rowLocal*H + k]     = *(uint4*)&h[0];
    *(uint4*)&Ahi[rowLocal*H + k + 8] = *(uint4*)&h[8];
    *(uint4*)&Alo[rowLocal*H + k]     = *(uint4*)&l[0];
    *(uint4*)&Alo[rowLocal*H + k + 8] = *(uint4*)&l[8];
  }
  if (tid < 64) { run[tid*2] = -__builtin_inff(); run[tid*2+1] = 0.0f; }
  __syncthreads();

  const int stageRow = tid >> 1;          // 0..127 (local column)
  const int stageK   = (tid & 1) * 16;    // 0 or 16

  for (int colBase = 0; colBase < nCols; colBase += 128) {
    v8f acc[4] = {};                      // 4 n-tiles of 16x16 fp32

    // 2-deep register prefetch of 16-float B sub-chunks
    float4 pre0[4], pre1[4];
    {
      const float4* s4 = (const float4*)(Cg + (size_t)(colBase + stageRow) * H + stageK);
      pre0[0] = s4[0]; pre0[1] = s4[1]; pre0[2] = s4[2]; pre0[3] = s4[3];
      const float4* s5 = (const float4*)(Cg + (size_t)(colBase + stageRow) * H + 32 + stageK);
      pre1[0] = s5[0]; pre1[1] = s5[1]; pre1[2] = s5[2]; pre1[3] = s5[3];
    }
    int p = 0;
    #pragma unroll 1
    for (int k0 = 0; k0 < H; k0 += 32) {
      bf16* BhiP = Bbuf + p * (2 * BCH);
      bf16* BloP = BhiP + BCH;
      // ---- convert + store current chunk into LDS buffer p ----
      {
        __align__(16) bf16 h[16]; __align__(16) bf16 l[16];
        #pragma unroll
        for (int q = 0; q < 4; ++q) {
          float4 f = pre0[q];
          split_bf16(f.x, h[4*q+0], l[4*q+0]);
          split_bf16(f.y, h[4*q+1], l[4*q+1]);
          split_bf16(f.z, h[4*q+2], l[4*q+2]);
          split_bf16(f.w, h[4*q+3], l[4*q+3]);
        }
        *(uint4*)&BhiP[stageRow*32 + stageK]     = *(uint4*)&h[0];
        *(uint4*)&BhiP[stageRow*32 + stageK + 8] = *(uint4*)&h[8];
        *(uint4*)&BloP[stageRow*32 + stageK]     = *(uint4*)&l[0];
        *(uint4*)&BloP[stageRow*32 + stageK + 8] = *(uint4*)&l[8];
      }
      // ---- rotate prefetch pipeline, issue load for k0+64 ----
      #pragma unroll
      for (int q = 0; q < 4; ++q) pre0[q] = pre1[q];
      if (k0 + 64 < H) {
        const float4* s4 = (const float4*)(Cg + (size_t)(colBase + stageRow) * H + k0 + 64 + stageK);
        pre1[0] = s4[0]; pre1[1] = s4[1]; pre1[2] = s4[2]; pre1[3] = s4[3];
      }
      __syncthreads();   // single barrier per k-step (double-buffered)

      // ---- A fragments per ISA 16-bit A 16x32 layout ----
      // lane<16: row=l15, K = {k0+0..7, k0+16..23}; lane>=16: K = {k0+8..15, k0+24..31}
      union { uint4 u[2]; v16bf v; } fa, fal;
      const bf16* pa  = &Ahi[(mt*16 + l15)*H + k0 + half*8];
      const bf16* pal = &Alo[(mt*16 + l15)*H + k0 + half*8];
      fa.u[0]  = *(const uint4*)pa;    fa.u[1]  = *(const uint4*)(pa  + 16);
      fal.u[0] = *(const uint4*)pal;   fal.u[1] = *(const uint4*)(pal + 16);

      #pragma unroll
      for (int j = 0; j < 4; ++j) {
        const int nt = ng*4 + j;
        // B 32x16: lane holds column N=l15, K = {half*8..half*8+7, 16+half*8..16+half*8+7}
        union { uint4 u[2]; v16bf v; } fb, fbl;
        const bf16* pb  = &BhiP[(nt*16 + l15)*32 + half*8];
        const bf16* pbl = &BloP[(nt*16 + l15)*32 + half*8];
        fb.u[0]  = *(const uint4*)pb;   fb.u[1]  = *(const uint4*)(pb  + 16);
        fbl.u[0] = *(const uint4*)pbl;  fbl.u[1] = *(const uint4*)(pbl + 16);
        acc[j] = wmma_bf16(fa.v,  fb.v,  acc[j]);  // hi*hi
        acc[j] = wmma_bf16(fa.v,  fbl.v, acc[j]);  // hi*lo
        acc[j] = wmma_bf16(fal.v, fb.v,  acc[j]);  // lo*hi
      }
      p ^= 1;
    }

    // ---- bias per column, then per-row (m,s) over this 128-col block ----
    float bv[4];
    #pragma unroll
    for (int j = 0; j < 4; ++j)
      bv[j] = bp ? bp[colBase + (ng*4 + j)*16 + l15] : 0.0f;

    #pragma unroll
    for (int r = 0; r < 8; ++r) {
      // C/D layout: VGPR r, lane<16 -> row r (cols=lane); lane>=16 -> row 8+r
      float v0 = acc[0][r] + bv[0];
      float v1 = acc[1][r] + bv[1];
      float v2 = acc[2][r] + bv[2];
      float v3 = acc[3][r] + bv[3];
      float m = fmaxf(fmaxf(v0, v1), fmaxf(v2, v3));
      float s = __expf(v0 - m) + __expf(v1 - m) + __expf(v2 - m) + __expf(v3 - m);
      { float m2 = swz_xor<1>(m), s2 = swz_xor<1>(s); lse_merge(m, s, m2, s2); }
      { float m2 = swz_xor<2>(m), s2 = swz_xor<2>(s); lse_merge(m, s, m2, s2); }
      { float m2 = swz_xor<4>(m), s2 = swz_xor<4>(s); lse_merge(m, s, m2, s2); }
      { float m2 = swz_xor<8>(m), s2 = swz_xor<8>(s); lse_merge(m, s, m2, s2); }
      if (l15 == 0) {
        int lr = half*8 + r;
        part[(w*16 + lr)*2]     = m;
        part[(w*16 + lr)*2 + 1] = s;
      }
    }
    __syncthreads();
    if (tid < 64) {  // merge the two n-groups + running state; tid = panel row
      int mtq = tid >> 4, rl = tid & 15;
      float m1 = part[((mtq    )*16 + rl)*2], s1 = part[((mtq    )*16 + rl)*2 + 1];
      float m2 = part[((mtq + 4)*16 + rl)*2], s2 = part[((mtq + 4)*16 + rl)*2 + 1];
      lse_merge(m1, s1, m2, s2);
      float rm = run[tid*2], rs = run[tid*2+1];
      lse_merge(rm, rs, m1, s1);
      run[tid*2] = rm; run[tid*2+1] = rs;
    }
    __syncthreads();
  }

  if (tid < 64) {
    float m = run[tid*2], s = run[tid*2+1];
    outp[rowBase + tid] = m + __logf(s);
  }
}

// ---------------------------------------------------------------------------
// Exact fp32 per-row dots: pos[r] = <E[r],X[r]>, tok[r] = <X[r],W[tgt]> + b[tgt]
// One wave per row.
// ---------------------------------------------------------------------------
__global__ __launch_bounds__(256) void rowdot_kernel(
    const float* __restrict__ X, const float* __restrict__ E,
    const float* __restrict__ Wv, const float* __restrict__ bias,
    const int* __restrict__ target,
    float* __restrict__ pos, float* __restrict__ tok)
{
  constexpr int H = 1024;
  const int lane = threadIdx.x & 31;
  const int w    = threadIdx.x >> 5;
  const int row  = blockIdx.x * 8 + w;
  const int tgt  = target[row];
  const float4* x4 = (const float4*)(X  + (size_t)row * H);
  const float4* e4 = (const float4*)(E  + (size_t)row * H);
  const float4* w4 = (const float4*)(Wv + (size_t)tgt * H);
  float dXE = 0.f, dXW = 0.f;
  #pragma unroll
  for (int i = 0; i < 8; ++i) {
    float4 xv = x4[lane + 32*i];
    float4 ev = e4[lane + 32*i];
    float4 wv = w4[lane + 32*i];
    dXE += xv.x*ev.x + xv.y*ev.y + xv.z*ev.z + xv.w*ev.w;
    dXW += xv.x*wv.x + xv.y*wv.y + xv.z*wv.z + xv.w*wv.w;
  }
  dXE = wave_sum32(dXE);
  dXW = wave_sum32(dXW);
  if (lane == 0) {
    pos[row] = dXE;
    tok[row] = dXW + bias[tgt];
  }
}

// ---------------------------------------------------------------------------
// Final scalar losses.
// ---------------------------------------------------------------------------
__global__ __launch_bounds__(256) void final_kernel(
    const float* __restrict__ tok, const float* __restrict__ lseM,
    const float* __restrict__ pos, const float* __restrict__ lseN,
    const int* __restrict__ masked_target, float* __restrict__ out,
    int rows, int mask_id)
{
  __shared__ float sA[256], sB[256], sC[256];
  float a = 0.f, bsum = 0.f, c = 0.f;
  for (int r = threadIdx.x; r < rows; r += 256) {
    a += tok[r] - lseM[r];
    if (masked_target[r] == mask_id) { bsum += pos[r] - lseN[r]; c += 1.f; }
  }
  sA[threadIdx.x] = a; sB[threadIdx.x] = bsum; sC[threadIdx.x] = c;
  __syncthreads();
  for (int st = 128; st > 0; st >>= 1) {
    if (threadIdx.x < st) {
      sA[threadIdx.x] += sA[threadIdx.x + st];
      sB[threadIdx.x] += sB[threadIdx.x + st];
      sC[threadIdx.x] += sC[threadIdx.x + st];
    }
    __syncthreads();
  }
  if (threadIdx.x == 0) {
    out[0] = -sA[0] / (float)rows;   // mim_loss
    out[1] = -sB[0] / sC[0];         // nce_loss
  }
}

extern "C" void kernel_launch(void* const* d_in, const int* in_sizes, int n_in,
                              void* d_out, int out_size, void* d_ws, size_t ws_size,
                              hipStream_t stream) {
  const float* X      = (const float*)d_in[0];  // hidden_states [8192,1024]
  const float* E      = (const float*)d_in[1];  // embedding     [8192,1024]
  const int*   target = (const int*)d_in[2];    // [8192]
  const int*   mtgt   = (const int*)d_in[3];    // [8192]
  const float* Wv     = (const float*)d_in[4];  // [8192,1024]
  const float* b      = (const float*)d_in[5];  // [8192]

  const int rows = in_sizes[2];   // B*S = 8192
  const int V    = in_sizes[5];   // 8192

  float* ws   = (float*)d_ws;
  float* lseM = ws;
  float* lseN = ws + rows;
  float* pos  = ws + 2 * rows;
  float* tok  = ws + 3 * rows;

  // LDS: A panel hi/lo (256KB) + double-buffered B chunk hi/lo (32KB) + state
  constexpr size_t SMEM = (size_t)64*1024*2*2 + (size_t)4*128*32*2
                        + (8*16*2 + 64*2) * sizeof(float);   // 296448 B < 320KB

  dim3 grid(rows / 64, 2);   // y=0 -> MIM (X·Wt+b), y=1 -> NCE (E·Xt)
  lse_gemm_kernel<<<grid, dim3(256), SMEM, stream>>>(X, Wv, b, lseM, E, lseN, V, rows);
  rowdot_kernel<<<rows / 8, 256, 0, stream>>>(X, E, Wv, b, target, pos, tok);
  final_kernel<<<1, 256, 0, stream>>>(tok, lseM, pos, lseN, mtgt, (float*)d_out, rows, 8192);
}
